// RicciCurvaturePooling_36962488550043
// MI455X (gfx1250) — compile-verified, matrix operand order checked by reading, not run on hardware
//
#include <hip/hip_runtime.h>
#include <hip/hip_bf16.h>

#define NEG_SLOPE 0.2f
#define EPS_F 1e-8f

typedef __attribute__((ext_vector_type(2)))  float    v2f;
typedef __attribute__((ext_vector_type(8)))  float    v8f;
typedef __attribute__((ext_vector_type(16))) _Float16 v16h;

#if __has_builtin(__builtin_amdgcn_wmma_f32_16x16x4_f32)
#define HAVE_WMMA_F32X4 1
#endif

#define CDIV(a,b) (((a)+(b)-1)/(b))

// ---------------- utility kernels ----------------
__global__ void rcp_fill_kernel(float* p, float v, long long n) {
  long long i = (long long)blockIdx.x * blockDim.x + threadIdx.x;
  if (i < n) p[i] = v;
}

__global__ void rcp_init_bias_kernel(float* p, const float* __restrict__ b, int nC) {
  int i = blockIdx.x * blockDim.x + threadIdx.x;
  if (i < nC) p[i] = b[i & 63];
}

// deg init to 1.0 (the extra self-loop added inside _gcn), then +1 per dst entry
__global__ void rcp_deg_accum_kernel(float* deg, const int* __restrict__ dstArr, int m) {
  int i = blockIdx.x * blockDim.x + threadIdx.x;
  if (i < m) atomicAdd(&deg[dstArr[i]], 1.0f);
}

__global__ void rcp_deg_to_dinv_kernel(float* deg, int n) {
  int i = blockIdx.x * blockDim.x + threadIdx.x;
  if (i < n) {
    float d = deg[i];
    deg[i] = rsqrtf(fmaxf(d, 1e-12f));
  }
}

// ---------------- WMMA fp32 GEMM: C[M,N] = A[M,K] @ B[K,N] (row major) ----------------
// One wave32 per 16x16 output tile. M,N multiples of 16; K multiple of 4 (or 32 fallback).
__global__ void rcp_wmma_gemm_kernel(const float* __restrict__ A, const float* __restrict__ B,
                                     float* __restrict__ C, int M, int N, int K) {
  const int ntn  = N >> 4;
  const int tile = blockIdx.x;
  const int tm = tile / ntn, tn = tile % ntn;
  const int lane = threadIdx.x & 31;
  const int half = lane >> 4;      // lanes 16-31 hold the upper K pair / upper M rows
  const int l16  = lane & 15;
  v8f acc = {0.f,0.f,0.f,0.f,0.f,0.f,0.f,0.f};
#ifdef HAVE_WMMA_F32X4
  // A 16x4 layout: lane<16 -> K={k0,k0+1}; lane>=16 -> K={k0+2,k0+3}; row M = l16
  const float* arow = A + (size_t)(tm * 16 + l16) * K + half * 2;
  const float* bcol = B + (size_t)(half * 2) * N + tn * 16 + l16;
  for (int k0 = 0; k0 < K; k0 += 4) {
    v2f a, b;
    a.x = arow[k0];
    a.y = arow[k0 + 1];
    b.x = bcol[(size_t)k0 * N];
    b.y = bcol[(size_t)(k0 + 1) * N];
    acc = __builtin_amdgcn_wmma_f32_16x16x4_f32(false, a, false, b, (short)0, acc, false, false);
  }
#else
  // Fallback: fp16 inputs, fp32 accumulate (codegen-confirmed builtin). K multiple of 32.
  const float* arow = A + (size_t)(tm * 16 + l16) * K;
  const float* bcol = B + tn * 16 + l16;
  for (int k0 = 0; k0 < K; k0 += 32) {
    v16h a, b;
#pragma unroll
    for (int j = 0; j < 8; ++j) {
      a[j]     = (_Float16)arow[k0 + half * 8 + j];
      a[8 + j] = (_Float16)arow[k0 + 16 + half * 8 + j];
      b[j]     = (_Float16)bcol[(size_t)(k0 + half * 8 + j) * N];
      b[8 + j] = (_Float16)bcol[(size_t)(k0 + 16 + half * 8 + j) * N];
    }
    acc = __builtin_amdgcn_wmma_f32_16x16x32_f16(false, a, false, b, (short)0, acc, false, false);
  }
#endif
  // C/D layout: VGPR r -> row tm*16 + half*8 + r, col tn*16 + l16
  float* crow = C + (size_t)(tm * 16 + half * 8) * N + tn * 16 + l16;
#pragma unroll
  for (int r = 0; r < 8; ++r) crow[(size_t)r * N] = acc[r];
}

// ---------------- GCN scatter: xpj[d] += xw[s] * dinv[s]*dinv[d] ----------------
// i in [0, e1): (src[i], dst[i]); i in [e1, e1+n): (i-e1, i-e1)  (extra self-loop)
__global__ void rcp_gcn_scatter_kernel(const float* __restrict__ xw, const float* __restrict__ dinv,
                                       const int* __restrict__ src, const int* __restrict__ dst,
                                       float* xpj, int e1, int n) {
  long long tid = (long long)blockIdx.x * blockDim.x + threadIdx.x;
  long long total = (long long)(e1 + n) * 64;
  if (tid >= total) return;
  int c = (int)(tid & 63);
  int i = (int)(tid >> 6);
  int s, d;
  if (i < e1) { s = src[i]; d = dst[i]; } else { s = d = i - e1; }
  float nrm = dinv[s] * dinv[d];
  atomicAdd(&xpj[(size_t)d * 64 + c], xw[(size_t)s * 64 + c] * nrm);
}

// ---------------- float atomic max via CAS ----------------
__device__ inline void rcp_atomicMaxF(float* addr, float val) {
  unsigned* ua = (unsigned*)addr;
  unsigned old = __float_as_uint(*addr);
  while (__uint_as_float(old) < val) {
    unsigned assumed = old;
    old = atomicCAS(ua, assumed, __float_as_uint(val));
    if (old == assumed) break;
  }
}

// hmax[sdst] = max over incoming of h[ssrc]  (per head*channel = 384 values)
__global__ void rcp_seg_max_kernel(const float* __restrict__ h, const int* __restrict__ ssrc,
                                   const int* __restrict__ sdst, float* hmax, int e1) {
  long long tid = (long long)blockIdx.x * blockDim.x + threadIdx.x;
  long long total = (long long)e1 * 384;
  if (tid >= total) return;
  int v = (int)(tid % 384);
  int i = (int)(tid / 384);
  rcp_atomicMaxF(&hmax[(size_t)sdst[i] * 384 + v], h[(size_t)ssrc[i] * 384 + v]);
}

// alpha = leakyrelu( hi . att_i + hmax[dst] . att_j ); amax = segmax(alpha)
__global__ void rcp_alpha_kernel(const float* __restrict__ h, const float* __restrict__ hmax,
                                 const float* __restrict__ att, const int* __restrict__ ssrc,
                                 const int* __restrict__ sdst, float* alpha, float* amax, int e1) {
  int tid = blockIdx.x * blockDim.x + threadIdx.x;
  if (tid >= e1 * 6) return;
  int hh = tid % 6, i = tid / 6;
  int s = ssrc[i], d = sdst[i];
  const float* hi = h + ((size_t)s * 6 + hh) * 64;
  const float* hm = hmax + ((size_t)d * 6 + hh) * 64;
  const float* ai = att + hh * 128;
  const float* aj = ai + 64;
  float acc = 0.f;
  for (int c = 0; c < 64; ++c) acc += hi[c] * ai[c] + hm[c] * aj[c];
  acc = (acc >= 0.f) ? acc : NEG_SLOPE * acc;
  alpha[tid] = acc;
  rcp_atomicMaxF(&amax[d * 6 + hh], acc);
}

// e = exp(alpha - amax[dst]); denom += e
__global__ void rcp_softexp_kernel(float* alpha, const float* __restrict__ amax, float* denom,
                                   const int* __restrict__ sdst, int e1) {
  int tid = blockIdx.x * blockDim.x + threadIdx.x;
  if (tid >= e1 * 6) return;
  int hh = tid % 6, i = tid / 6;
  int d = sdst[i];
  float e = expf(alpha[tid] - amax[d * 6 + hh]);
  alpha[tid] = e;
  atomicAdd(&denom[d * 6 + hh], e);
}

// xa[dst] += h[src] * (e / denom[dst])
__global__ void rcp_xa_scatter_kernel(const float* __restrict__ h, const float* __restrict__ alpha,
                                      const float* __restrict__ denom, const int* __restrict__ ssrc,
                                      const int* __restrict__ sdst, float* xa, int e1) {
  long long tid = (long long)blockIdx.x * blockDim.x + threadIdx.x;
  long long total = (long long)e1 * 384;
  if (tid >= total) return;
  int c = (int)(tid & 63);
  long long t2 = tid >> 6;
  int hh = (int)(t2 % 6);
  int i  = (int)(t2 / 6);
  int s = ssrc[i], d = sdst[i];
  float coef = alpha[i * 6 + hh] / denom[d * 6 + hh];
  atomicAdd(&xa[((size_t)d * 6 + hh) * 64 + c], h[((size_t)s * 6 + hh) * 64 + c] * coef);
}

__global__ void rcp_head_mean_kernel(const float* __restrict__ xa, float* xm, int n) {
  int tid = blockIdx.x * blockDim.x + threadIdx.x;
  if (tid >= n * 64) return;
  int c = tid & 63, node = tid >> 6;
  float s = 0.f;
  for (int hh = 0; hh < 6; ++hh) s += xa[((size_t)node * 6 + hh) * 64 + c];
  xm[tid] = s * (1.0f / 6.0f);
}

__global__ void rcp_bsc_kernel(const float* __restrict__ xm, const float* __restrict__ W2,
                               const float* __restrict__ b2, float* bsc, int n) {
  int i = blockIdx.x * blockDim.x + threadIdx.x;
  if (i >= n) return;
  float s = b2[0];
  for (int c = 0; c < 64; ++c) s += xm[(size_t)i * 64 + c] * W2[c];
  bsc[i] = s;
}

__global__ void rcp_score_kernel(const float* __restrict__ a, const float* __restrict__ bsc,
                                 const int* __restrict__ ssrc, const int* __restrict__ sdst,
                                 float* score, int e1) {
  int i = blockIdx.x * blockDim.x + threadIdx.x;
  if (i >= e1) return;
  int s = ssrc[i], d = sdst[i];
  const float* ad = a + (size_t)d * 64;
  const float* as = a + (size_t)s * 64;
  float dij = 0.f, nd = 0.f, ns = 0.f;
  for (int c = 0; c < 64; ++c) {
    float u = ad[c], v = as[c];
    dij += u * v; nd += u * u; ns += v * v;
  }
  float cosv = dij / (fmaxf(sqrtf(nd), EPS_F) * fmaxf(sqrtf(ns), EPS_F));
  atomicAdd(&score[d], bsc[d] * cosv);
}

__global__ void rcp_sigmoid_kernel(const float* __restrict__ score, float* out, int n) {
  int i = blockIdx.x * blockDim.x + threadIdx.x;
  if (i < n) out[i] = 1.0f / (1.0f + expf(-score[i]));
}

// ---------------- launch ----------------
extern "C" void kernel_launch(void* const* d_in, const int* in_sizes, int n_in,
                              void* d_out, int out_size, void* d_ws, size_t ws_size,
                              hipStream_t stream) {
  const float* x      = (const float*)d_in[0];
  const float* W_gcn  = (const float*)d_in[1];
  const float* b_gcn  = (const float*)d_in[2];
  const float* weight = (const float*)d_in[3];
  const float* att    = (const float*)d_in[4];
  const float* W1     = (const float*)d_in[5];
  const float* W2     = (const float*)d_in[6];
  const float* b2     = (const float*)d_in[7];
  const int* src      = (const int*)d_in[8];
  const int* dst      = (const int*)d_in[9];
  const int* ssrc     = (const int*)d_in[10];
  const int* sdst     = (const int*)d_in[11];

  const int C_ = 64, H_ = 6;
  const int n  = in_sizes[0] / C_;   // 50000
  const int e1 = in_sizes[8];        // 450000 = E + N (includes one set of self-loops)

  // workspace carve-out (floats), with reuse
  float* W = (float*)d_ws;
  size_t o = 0;
  float* hbuf  = W + o; o += (size_t)n * H_ * C_;   // h = xpj @ weight          (76.8MB)
  float* hmax  = W + o; o += (size_t)n * H_ * C_;   // segment max -> reused as xa
  float* xw    = W + o; o += (size_t)n * C_;        // x @ W_gcn -> reused as xm
  float* xpj   = W + o; o += (size_t)n * C_;        // gcn output -> reused as a
  float* alpha = W + o; o += (size_t)e1 * H_;       // per-edge attention logits/exp
  float* amax  = W + o; o += (size_t)n * H_;
  float* denom = W + o; o += (size_t)n * H_;
  float* dinv  = W + o; o += (size_t)n;
  float* bsc   = W + o; o += (size_t)n;
  float* score = W + o; o += (size_t)n;
  (void)ws_size; (void)n_in;

  const float NEG_INF = -__builtin_inff();
  const int B = 256;

  // 1) degrees -> dinv   (deg = 1 self-loop + counts over dst array)
  rcp_fill_kernel<<<CDIV((long long)n, B), B, 0, stream>>>(dinv, 1.0f, n);
  rcp_deg_accum_kernel<<<CDIV(e1, B), B, 0, stream>>>(dinv, dst, e1);
  rcp_deg_to_dinv_kernel<<<CDIV(n, B), B, 0, stream>>>(dinv, n);

  // 2) xw = x @ W_gcn   (50000x64 @ 64x64)
  rcp_wmma_gemm_kernel<<<(n / 16) * (C_ / 16), 32, 0, stream>>>(x, W_gcn, xw, n, C_, C_);

  // 3) xpj = bias + segment_sum(xw[s] * norm)
  rcp_init_bias_kernel<<<CDIV(n * C_, B), B, 0, stream>>>(xpj, b_gcn, n * C_);
  {
    long long total = (long long)(e1 + n) * 64;
    rcp_gcn_scatter_kernel<<<(unsigned)CDIV(total, (long long)B), B, 0, stream>>>(
        xw, dinv, src, dst, xpj, e1, n);
  }

  // 4) h = xpj @ weight  (50000x64 @ 64x384)
  rcp_wmma_gemm_kernel<<<(n / 16) * ((H_ * C_) / 16), 32, 0, stream>>>(xpj, weight, hbuf, n, H_ * C_, C_);

  // 5) hmax = segment_max(h[ssrc], sdst)
  rcp_fill_kernel<<<CDIV((long long)n * H_ * C_, B), B, 0, stream>>>(hmax, NEG_INF, (long long)n * H_ * C_);
  {
    long long total = (long long)e1 * 384;
    rcp_seg_max_kernel<<<(unsigned)CDIV(total, (long long)B), B, 0, stream>>>(hbuf, ssrc, sdst, hmax, e1);
  }

  // 6) attention logits + segment max
  rcp_fill_kernel<<<CDIV((long long)n * H_, B), B, 0, stream>>>(amax, NEG_INF, (long long)n * H_);
  rcp_alpha_kernel<<<CDIV(e1 * H_, B), B, 0, stream>>>(hbuf, hmax, att, ssrc, sdst, alpha, amax, e1);

  // 7) softmax exp + denom
  rcp_fill_kernel<<<CDIV((long long)n * H_, B), B, 0, stream>>>(denom, 0.0f, (long long)n * H_);
  rcp_softexp_kernel<<<CDIV(e1 * H_, B), B, 0, stream>>>(alpha, amax, denom, sdst, e1);

  // 8) xa = segment_sum(h[ssrc] * alpha)   (xa aliases hmax buffer; hmax is dead now)
  float* xa = hmax;
  rcp_fill_kernel<<<CDIV((long long)n * H_ * C_, B), B, 0, stream>>>(xa, 0.0f, (long long)n * H_ * C_);
  {
    long long total = (long long)e1 * 384;
    rcp_xa_scatter_kernel<<<(unsigned)CDIV(total, (long long)B), B, 0, stream>>>(
        hbuf, alpha, denom, ssrc, sdst, xa, e1);
  }

  // 9) xm = mean over heads   (xm aliases xw)
  float* xm = xw;
  rcp_head_mean_kernel<<<CDIV(n * C_, B), B, 0, stream>>>(xa, xm, n);

  // 10) a = xm @ W1 (aliases xpj); bsc = xm @ W2 + b2
  float* a = xpj;
  rcp_wmma_gemm_kernel<<<(n / 16) * (C_ / 16), 32, 0, stream>>>(xm, W1, a, n, C_, C_);
  rcp_bsc_kernel<<<CDIV(n, B), B, 0, stream>>>(xm, W2, b2, bsc, n);

  // 11) score = segment_sum(bsc[dst] * cos(a[dst], a[src]))
  rcp_fill_kernel<<<CDIV((long long)n, B), B, 0, stream>>>(score, 0.0f, n);
  rcp_score_kernel<<<CDIV(e1, B), B, 0, stream>>>(a, bsc, ssrc, sdst, score, e1);

  // 12) fitness = sigmoid(score)
  rcp_sigmoid_kernel<<<CDIV(n, B), B, 0, stream>>>(score, (float*)d_out, n);
}
